// EncLayer_13864154431833
// MI455X (gfx1250) — matrix operand order repaired and checked by reading, not compiled, not run on hardware
//
#include <hip/hip_runtime.h>
#include <hip/hip_bf16.h>
#include <math.h>

#define L_NODES 4096
#define K_NB    48
#define H_DIM   128

typedef __attribute__((ext_vector_type(16))) __bf16 v16bf;
typedef __attribute__((ext_vector_type(8)))  float  v8f;
typedef __attribute__((ext_vector_type(8)))  unsigned int v8u;

// ---------- helpers ----------

__device__ __forceinline__ float gelu_exact(float x) {
    return 0.5f * x * (1.0f + erff(x * 0.70710678118654752f));
}

__device__ __forceinline__ float wave_red_sum(float x) {
#pragma unroll
    for (int o = 16; o > 0; o >>= 1) x += __shfl_xor(x, o, 32);
    return x;
}

// Load a 16x32 bf16 A-fragment from LDS (row-major, `stride` in elements).
// Per CDNA5 ISA 7.12.2: lanes 0-15 row M, VGPR0-3 = K pairs 0..7, VGPR4-7 = K 16..23;
// lanes 16-31 get the +8 K offset. All offsets even -> 8 x ds_load_b32.
__device__ __forceinline__ v16bf lds_a_frag(const __bf16* base, int stride,
                                            int mbase, int kbase, int lane) {
    int m  = mbase + (lane & 15);
    int k0 = kbase + ((lane >> 4) << 3);
    const unsigned int* p = (const unsigned int*)(base + m * stride + k0);
    v8u t;
#pragma unroll
    for (int v = 0; v < 8; ++v) t[v] = p[(v & 3) + ((v >> 2) << 3)];
    return __builtin_bit_cast(v16bf, t);
}

__device__ __forceinline__ v8f wmma_bf16(v16bf a, v16bf b, v8f c) {
    return __builtin_amdgcn_wmma_f32_16x16x32_bf16(false, a, false, b,
                                                   (short)0, c, false, false);
}

// ---------- weight swizzle: f32 [K,N] -> bf16 WMMA B-fragment order ----------
// Fragment block = (ntile * ksteps + kstep), 32 lanes x 16 halfs contiguous.
// B layout (32x16 bf16): lanes 0-15 hold K=0..7,16..23 (2 per dword), lanes 16-31 K=8..15,24..31.
__global__ __launch_bounds__(256) void swizzle_w(const float* __restrict__ W,
                                                 __bf16* __restrict__ dst,
                                                 int K, int N) {
    int idx = blockIdx.x * 256 + threadIdx.x;
    if (idx >= K * N) return;
    int e    = idx & 15;
    int lane = (idx >> 4) & 31;
    int blk  = idx >> 9;
    int ksteps = K >> 5;
    int nt = blk / ksteps;
    int ks = blk - nt * ksteps;
    int v = e >> 1, hi = e & 1;
    int kk = ks * 32 + ((lane >> 4) << 3) + ((v >> 2) << 4) + 2 * (v & 3) + hi;
    int n  = nt * 16 + (lane & 15);
    dst[idx] = (__bf16)W[kk * N + n];
}

// ---------- kernel A: node message MLP + aggregate + LN1 ----------
__global__ __launch_bounds__(256) void node_msg_kernel(
    const float* __restrict__ hV, const float* __restrict__ hE,
    const int* __restrict__ Eidx, const float* __restrict__ maskAtt,
    const v16bf* __restrict__ W1s, const float* __restrict__ b1,
    const v16bf* __restrict__ W2s, const float* __restrict__ b2,
    const v16bf* __restrict__ W3s, const float* __restrict__ b3,
    const float* __restrict__ g1, const float* __restrict__ bb1,
    float* __restrict__ hv_out) {
    const int i = blockIdx.x;
    const int tid = threadIdx.x;
    const int wave = tid >> 5, lane = tid & 31;

    __shared__ __align__(16) __bf16 regionA[48 * 388];   // A matrix, later B2
    __shared__ __align__(16) float  regionB[48 * 130];   // B1 (bf16 view)
    __shared__ float hv_i[128];
    __shared__ float dh[128];
    __shared__ float maskA[48];
    __shared__ int   nidx[48];
    __shared__ float s_mean, s_rstd;

    for (int c = tid; c < 128; c += 256) hv_i[c] = hV[i * 128 + c];
    if (tid < 48) { maskA[tid] = maskAtt[i * 48 + tid]; nidx[tid] = Eidx[i * 48 + tid]; }
    __syncthreads();

    // Build A = [hV_i | hE_row | hV[nbr]]  -> [48, 384] bf16
    for (int idx = tid; idx < 48 * 384; idx += 256) {
        int r = idx / 384, c = idx - r * 384;
        float v;
        if (c < 128)      v = hv_i[c];
        else if (c < 256) v = hE[(i * 48 + r) * 128 + (c - 128)];
        else              v = hV[nidx[r] * 128 + (c - 256)];
        regionA[r * 388 + c] = (__bf16)v;
    }
    __syncthreads();

    __bf16* B1 = (__bf16*)regionB;       // [48,128] stride 136
    {   // GEMM1: [48,384]x[384,128], gelu
        const int col = wave * 16 + (lane & 15);
        const float bias = b1[col];
#pragma unroll
        for (int mt = 0; mt < 3; ++mt) {
            v8f acc = {};
            for (int ks = 0; ks < 12; ++ks) {
                v16bf a = lds_a_frag(regionA, 388, mt * 16, ks * 32, lane);
                v16bf b = W1s[(wave * 12 + ks) * 32 + lane];
                acc = wmma_bf16(a, b, acc);
            }
#pragma unroll
            for (int v = 0; v < 8; ++v) {
                int m = mt * 16 + v + ((lane >> 4) << 3);
                B1[m * 136 + col] = (__bf16)gelu_exact(acc[v] + bias);
            }
        }
    }
    __syncthreads();
    __bf16* B2 = regionA;                // [48,128] stride 136 (A is dead)
    {   // GEMM2: [48,128]x[128,128], gelu
        const int col = wave * 16 + (lane & 15);
        const float bias = b2[col];
#pragma unroll
        for (int mt = 0; mt < 3; ++mt) {
            v8f acc = {};
            for (int ks = 0; ks < 4; ++ks) {
                v16bf a = lds_a_frag(B1, 136, mt * 16, ks * 32, lane);
                v16bf b = W2s[(wave * 4 + ks) * 32 + lane];
                acc = wmma_bf16(a, b, acc);
            }
#pragma unroll
            for (int v = 0; v < 8; ++v) {
                int m = mt * 16 + v + ((lane >> 4) << 3);
                B2[m * 136 + col] = (__bf16)gelu_exact(acc[v] + bias);
            }
        }
    }
    __syncthreads();
    {   // GEMM3 + masked sum over neighbors (rows) straight from accumulators
        const int col = wave * 16 + (lane & 15);
        const float bias = b3[col];
        float s = 0.f;
#pragma unroll
        for (int mt = 0; mt < 3; ++mt) {
            v8f acc = {};
            for (int ks = 0; ks < 4; ++ks) {
                v16bf a = lds_a_frag(B2, 136, mt * 16, ks * 32, lane);
                v16bf b = W3s[(wave * 4 + ks) * 32 + lane];
                acc = wmma_bf16(a, b, acc);
            }
#pragma unroll
            for (int v = 0; v < 8; ++v) {
                int m = mt * 16 + v + ((lane >> 4) << 3);
                s += maskA[m] * (acc[v] + bias);
            }
        }
        s += __shfl_xor(s, 16, 32);      // lanes l and l+16 hold same column
        if (lane < 16) dh[col] = s * (1.0f / 30.0f);
    }
    __syncthreads();
    if (wave == 0) {                     // LN stats over 128 features
        float sum = 0.f, sq = 0.f;
#pragma unroll
        for (int j = 0; j < 4; ++j) {
            int c = lane * 4 + j;
            float x = hv_i[c] + dh[c];
            sum += x; sq += x * x;
        }
        sum = wave_red_sum(sum); sq = wave_red_sum(sq);
        if (lane == 0) {
            float mean = sum * (1.f / 128.f);
            float var  = sq * (1.f / 128.f) - mean * mean;
            s_mean = mean;
            s_rstd = rsqrtf(var + 1e-5f);
        }
    }
    __syncthreads();
    if (tid < 128) {
        float x = hv_i[tid] + dh[tid];
        hv_out[i * 128 + tid] = g1[tid] * (x - s_mean) * s_rstd + bb1[tid];
    }
}

// ---------- kernel B: position-wise FFN (16 nodes per block) + LN2 + mask ----------
__global__ __launch_bounds__(256) void ffn_kernel(
    float* __restrict__ hv,              // in/out: d_out node region
    const v16bf* __restrict__ Wis, const float* __restrict__ bi,
    const v16bf* __restrict__ Wos, const float* __restrict__ bo,
    const float* __restrict__ g2, const float* __restrict__ bb2,
    const float* __restrict__ maskV) {
    const int m0 = blockIdx.x * 16;
    const int tid = threadIdx.x;
    const int wave = tid >> 5, lane = tid & 31;

    __shared__ __align__(16) __bf16 Abf[16 * 136];
    __shared__ __align__(16) __bf16 Hbf[16 * 520];
    __shared__ float outF[16 * 132];

    for (int idx = tid; idx < 16 * 128; idx += 256) {
        int r = idx >> 7, c = idx & 127;
        Abf[r * 136 + c] = (__bf16)hv[(m0 + r) * 128 + c];
    }
    __syncthreads();
    // [16,128] x [128,512], gelu -> Hbf
#pragma unroll
    for (int j = 0; j < 4; ++j) {
        int nt = wave + 8 * j;
        int col = nt * 16 + (lane & 15);
        float bias = bi[col];
        v8f acc = {};
        for (int ks = 0; ks < 4; ++ks) {
            v16bf a = lds_a_frag(Abf, 136, 0, ks * 32, lane);
            v16bf b = Wis[(nt * 4 + ks) * 32 + lane];
            acc = wmma_bf16(a, b, acc);
        }
#pragma unroll
        for (int v = 0; v < 8; ++v) {
            int r = v + ((lane >> 4) << 3);
            Hbf[r * 520 + col] = (__bf16)gelu_exact(acc[v] + bias);
        }
    }
    __syncthreads();
    // [16,512] x [512,128] -> outF
    {
        int col = wave * 16 + (lane & 15);
        float bias = bo[col];
        v8f acc = {};
        for (int ks = 0; ks < 16; ++ks) {
            v16bf a = lds_a_frag(Hbf, 520, 0, ks * 32, lane);
            v16bf b = Wos[(wave * 16 + ks) * 32 + lane];
            acc = wmma_bf16(a, b, acc);
        }
#pragma unroll
        for (int v = 0; v < 8; ++v) {
            int r = v + ((lane >> 4) << 3);
            outF[r * 132 + col] = acc[v] + bias;
        }
    }
    __syncthreads();
    // residual + LN2 + mask_V; wave handles rows (wave, wave+8)
#pragma unroll
    for (int j = 0; j < 2; ++j) {
        int r = wave + 8 * j;
        int gm = m0 + r;
        float xv[4], sum = 0.f, sq = 0.f;
#pragma unroll
        for (int jj = 0; jj < 4; ++jj) {
            int c = lane * 4 + jj;
            float x = outF[r * 132 + c] + hv[gm * 128 + c];
            xv[jj] = x; sum += x; sq += x * x;
        }
        sum = wave_red_sum(sum); sq = wave_red_sum(sq);
        float mean = sum * (1.f / 128.f);
        float rstd = rsqrtf(sq * (1.f / 128.f) - mean * mean + 1e-5f);
        float mk = maskV[gm];
#pragma unroll
        for (int jj = 0; jj < 4; ++jj) {
            int c = lane * 4 + jj;
            hv[gm * 128 + c] = (g2[c] * (xv[jj] - mean) * rstd + bb2[c]) * mk;
        }
    }
}

// ---------- kernel C: edge update MLP + LN3 ----------
__global__ __launch_bounds__(256) void edge_upd_kernel(
    const float* __restrict__ hVn,       // updated h_V (d_out node region)
    const float* __restrict__ hE, const int* __restrict__ Eidx,
    const v16bf* __restrict__ W11s, const float* __restrict__ b11,
    const v16bf* __restrict__ W12s, const float* __restrict__ b12,
    const v16bf* __restrict__ W13s, const float* __restrict__ b13,
    const float* __restrict__ g3, const float* __restrict__ bb3,
    float* __restrict__ he_out) {
    const int i = blockIdx.x;
    const int tid = threadIdx.x;
    const int wave = tid >> 5, lane = tid & 31;

    __shared__ __align__(16) __bf16 regionA[48 * 388];
    __shared__ __align__(16) float  regionB[48 * 130];
    __shared__ float hv_i[128];
    __shared__ int   nidx[48];

    for (int c = tid; c < 128; c += 256) hv_i[c] = hVn[i * 128 + c];
    if (tid < 48) nidx[tid] = Eidx[i * 48 + tid];
    __syncthreads();

    for (int idx = tid; idx < 48 * 384; idx += 256) {
        int r = idx / 384, c = idx - r * 384;
        float v;
        if (c < 128)      v = hv_i[c];
        else if (c < 256) v = hE[(i * 48 + r) * 128 + (c - 128)];
        else              v = hVn[nidx[r] * 128 + (c - 256)];
        regionA[r * 388 + c] = (__bf16)v;
    }
    __syncthreads();

    __bf16* B1 = (__bf16*)regionB;
    {
        const int col = wave * 16 + (lane & 15);
        const float bias = b11[col];
#pragma unroll
        for (int mt = 0; mt < 3; ++mt) {
            v8f acc = {};
            for (int ks = 0; ks < 12; ++ks) {
                v16bf a = lds_a_frag(regionA, 388, mt * 16, ks * 32, lane);
                v16bf b = W11s[(wave * 12 + ks) * 32 + lane];
                acc = wmma_bf16(a, b, acc);
            }
#pragma unroll
            for (int v = 0; v < 8; ++v) {
                int m = mt * 16 + v + ((lane >> 4) << 3);
                B1[m * 136 + col] = (__bf16)gelu_exact(acc[v] + bias);
            }
        }
    }
    __syncthreads();
    __bf16* B2 = regionA;
    {
        const int col = wave * 16 + (lane & 15);
        const float bias = b12[col];
#pragma unroll
        for (int mt = 0; mt < 3; ++mt) {
            v8f acc = {};
            for (int ks = 0; ks < 4; ++ks) {
                v16bf a = lds_a_frag(B1, 136, mt * 16, ks * 32, lane);
                v16bf b = W12s[(wave * 4 + ks) * 32 + lane];
                acc = wmma_bf16(a, b, acc);
            }
#pragma unroll
            for (int v = 0; v < 8; ++v) {
                int m = mt * 16 + v + ((lane >> 4) << 3);
                B2[m * 136 + col] = (__bf16)gelu_exact(acc[v] + bias);
            }
        }
    }
    __syncthreads();
    float* outF = regionB;               // f32 [48,128] stride 130 (B1 dead)
    {
        const int col = wave * 16 + (lane & 15);
        const float bias = b13[col];
#pragma unroll
        for (int mt = 0; mt < 3; ++mt) {
            v8f acc = {};
            for (int ks = 0; ks < 4; ++ks) {
                v16bf a = lds_a_frag(B2, 136, mt * 16, ks * 32, lane);
                v16bf b = W13s[(wave * 4 + ks) * 32 + lane];
                acc = wmma_bf16(a, b, acc);
            }
#pragma unroll
            for (int v = 0; v < 8; ++v) {
                int m = mt * 16 + v + ((lane >> 4) << 3);
                outF[m * 130 + col] = acc[v] + bias + hE[(i * 48 + m) * 128 + col];
            }
        }
    }
    __syncthreads();
    // per-row LN3; wave handles rows wave, wave+8, ..., wave+40
#pragma unroll
    for (int j = 0; j < 6; ++j) {
        int r = wave + 8 * j;
        float xv[4], sum = 0.f, sq = 0.f;
#pragma unroll
        for (int jj = 0; jj < 4; ++jj) {
            int c = lane * 4 + jj;
            float x = outF[r * 130 + c];
            xv[jj] = x; sum += x; sq += x * x;
        }
        sum = wave_red_sum(sum); sq = wave_red_sum(sq);
        float mean = sum * (1.f / 128.f);
        float rstd = rsqrtf(sq * (1.f / 128.f) - mean * mean + 1e-5f);
#pragma unroll
        for (int jj = 0; jj < 4; ++jj) {
            int c = lane * 4 + jj;
            he_out[(i * 48 + r) * 128 + c] = g3[c] * (xv[jj] - mean) * rstd + bb3[c];
        }
    }
}

// ---------- launch ----------
extern "C" void kernel_launch(void* const* d_in, const int* in_sizes, int n_in,
                              void* d_out, int out_size, void* d_ws, size_t ws_size,
                              hipStream_t stream) {
    const float* hV      = (const float*)d_in[0];
    const float* hE      = (const float*)d_in[1];
    const int*   Eidx    = (const int*)d_in[2];
    const float* maskV   = (const float*)d_in[3];
    const float* maskAtt = (const float*)d_in[4];
    const float* W1_w  = (const float*)d_in[5];  const float* W1_b  = (const float*)d_in[6];
    const float* W2_w  = (const float*)d_in[7];  const float* W2_b  = (const float*)d_in[8];
    const float* W3_w  = (const float*)d_in[9];  const float* W3_b  = (const float*)d_in[10];
    const float* W11_w = (const float*)d_in[11]; const float* W11_b = (const float*)d_in[12];
    const float* W12_w = (const float*)d_in[13]; const float* W12_b = (const float*)d_in[14];
    const float* W13_w = (const float*)d_in[15]; const float* W13_b = (const float*)d_in[16];
    const float* Wi_w  = (const float*)d_in[17]; const float* Wi_b  = (const float*)d_in[18];
    const float* Wo_w  = (const float*)d_in[19]; const float* Wo_b  = (const float*)d_in[20];
    const float* n1_g  = (const float*)d_in[21]; const float* n1_b  = (const float*)d_in[22];
    const float* n2_g  = (const float*)d_in[23]; const float* n2_b  = (const float*)d_in[24];
    const float* n3_g  = (const float*)d_in[25]; const float* n3_b  = (const float*)d_in[26];

    float* hv_out = (float*)d_out;                       // [4096,128]
    float* he_out = hv_out + L_NODES * H_DIM;            // [4096,48,128]

    __bf16* ws = (__bf16*)d_ws;
    const size_t OFF_W1  = 0;        // 384x128 = 49152
    const size_t OFF_W2  = 49152;    // 128x128 = 16384
    const size_t OFF_W3  = 65536;
    const size_t OFF_W11 = 81920;
    const size_t OFF_W12 = 131072;
    const size_t OFF_W13 = 147456;
    const size_t OFF_WI  = 163840;   // 128x512 = 65536
    const size_t OFF_WO  = 229376;   // 512x128 = 65536

    auto nb = [](int n) { return (n + 255) / 256; };
    swizzle_w<<<nb(49152), 256, 0, stream>>>(W1_w,  ws + OFF_W1,  384, 128);
    swizzle_w<<<nb(16384), 256, 0, stream>>>(W2_w,  ws + OFF_W2,  128, 128);
    swizzle_w<<<nb(16384), 256, 0, stream>>>(W3_w,  ws + OFF_W3,  128, 128);
    swizzle_w<<<nb(49152), 256, 0, stream>>>(W11_w, ws + OFF_W11, 384, 128);
    swizzle_w<<<nb(16384), 256, 0, stream>>>(W12_w, ws + OFF_W12, 128, 128);
    swizzle_w<<<nb(16384), 256, 0, stream>>>(W13_w, ws + OFF_W13, 128, 128);
    swizzle_w<<<nb(65536), 256, 0, stream>>>(Wi_w,  ws + OFF_WI,  128, 512);
    swizzle_w<<<nb(65536), 256, 0, stream>>>(Wo_w,  ws + OFF_WO,  512, 128);

    node_msg_kernel<<<L_NODES, 256, 0, stream>>>(
        hV, hE, Eidx, maskAtt,
        (const v16bf*)(ws + OFF_W1), W1_b,
        (const v16bf*)(ws + OFF_W2), W2_b,
        (const v16bf*)(ws + OFF_W3), W3_b,
        n1_g, n1_b, hv_out);

    ffn_kernel<<<L_NODES / 16, 256, 0, stream>>>(
        hv_out,
        (const v16bf*)(ws + OFF_WI), Wi_b,
        (const v16bf*)(ws + OFF_WO), Wo_b,
        n2_g, n2_b, maskV);

    edge_upd_kernel<<<L_NODES, 256, 0, stream>>>(
        hv_out, hE, Eidx,
        (const v16bf*)(ws + OFF_W11), W11_b,
        (const v16bf*)(ws + OFF_W12), W12_b,
        (const v16bf*)(ws + OFF_W13), W13_b,
        n3_g, n3_b, he_out);

    (void)in_sizes; (void)n_in; (void)out_size; (void)ws_size;
}
